// DeepSpanLayer_6717328851360
// MI455X (gfx1250) — matrix operand order repaired and checked by reading, not compiled, not run on hardware
//
#include <hip/hip_runtime.h>
#include <hip/hip_bf16.h>

#define DIM 256
#define FFN3 1024
#define TSTEPS 32768
#define LN_EPS 1e-6f

typedef __attribute__((ext_vector_type(16))) __bf16 v16bf;
typedef __attribute__((ext_vector_type(8)))  float  v8f;
typedef __attribute__((ext_vector_type(4)))  float  f32x4;
typedef __attribute__((ext_vector_type(4)))  unsigned int u32x4;

__device__ __forceinline__ __bf16 f2bf(float x) { return (__bf16)x; }

// ---------------- elementwise f32 -> bf16 ----------------
__global__ void k_cvt_bf16(const float* __restrict__ in, __bf16* __restrict__ out, int n) {
    int i = blockIdx.x * blockDim.x + threadIdx.x;
    if (i < n) out[i] = f2bf(in[i]);
}

// ---------- transpose+convert: BT[n*K + k] = bf16(W[k*N + n]) ----------
__global__ void k_transpose_bf16(const float* __restrict__ W, __bf16* __restrict__ BT,
                                 int K, int N) {
    int i = blockIdx.x * blockDim.x + threadIdx.x;
    if (i >= K * N) return;
    int n = i / K, k = i % K;
    BT[i] = f2bf(W[(size_t)k * N + n]);
}

// ---------------- WMMA fragment loaders ----------------
// A row-major [M,K] bf16. ISA 16-bit A 16x32 layout: lanes0-15 row M=lane,
// K={0..7,16..23}; lanes16-31 same rows, K={8..15,24..31}; 2 K per VGPR.
__device__ __forceinline__ v16bf load_fragA(const __bf16* A, int K, int row, int k0, int lane) {
    int hi = (lane >> 4) & 1;
    const __bf16* p = A + (size_t)row * K + k0 + hi * 8;
    union { v16bf v; u32x4 q[2]; } u;
    u.q[0] = *(const u32x4*)(p);
    u.q[1] = *(const u32x4*)(p + 16);
    return u.v;
}
// B fragment from LDS chunk (64 cols x 256 K, bf16, col-major rows of 256).
// lanes0-15 col N=lane, K=k0+{0..7,16..23}; lanes16-31 same cols, K=k0+{8..15,24..31}.
__device__ __forceinline__ v16bf load_fragB_lds(const __bf16* bb, int col, int k) {
    const __bf16* p = bb + col * 256 + k;
    union { v16bf v; u32x4 q[2]; } u;
    u.q[0] = *(const u32x4*)(p);
    u.q[1] = *(const u32x4*)(p + 16);
    return u.v;
}
// cooperative stage of one 64x256 B chunk into LDS (32 KB), 256 threads
__device__ __forceinline__ void stageB(__bf16* dst, const __bf16* BT,
                                       int nBase, int kBase, int K, int tid) {
#pragma unroll
    for (int j = 0; j < 8; ++j) {
        int c = tid + j * 256;        // 0..2047 16-byte chunks
        int col = c >> 5;
        int ko = (c & 31) * 8;
        *(u32x4*)(dst + col * 256 + ko) =
            *(const u32x4*)(BT + (size_t)(nBase + col) * K + kBase + ko);
    }
}

// ---------------- bf16 WMMA GEMM: C[M,N] = A[M,K] x B[K,N] ----------------
// Block = 256 threads (8 waves). Wave tile 32x64 (8 WMMAs/k-step), block tile 256x64.
// B tile staged in LDS (double-buffered 256-K chunks); A register double-buffered.
// EPI 0: store f32.  EPI 1: +bias, relu, store bf16 (FFN layer 1).
template<int EPI>
__global__ void __launch_bounds__(256)
k_gemm_bf16(const __bf16* __restrict__ A, const __bf16* __restrict__ BT,
            const float* __restrict__ bias, float* __restrict__ Cf,
            __bf16* __restrict__ Cb, int M, int N, int K) {
    __shared__ __bf16 bsh[2][64 * 256];   // 2 x 32 KB
    const int tid = threadIdx.x;
    const int lane = tid & 31, wave = tid >> 5;
    const int mBase = blockIdx.x * 256 + wave * 32;
    const int nBase = blockIdx.y * 64;
    const int lo = lane & 15, hi = (lane >> 4) & 1;
    const int rowA0 = mBase + lo;
    const int rowA1 = mBase + 16 + lo;

    v8f zero = {0.f,0.f,0.f,0.f,0.f,0.f,0.f,0.f};
    v8f acc[2][4] = {{zero, zero, zero, zero}, {zero, zero, zero, zero}};

    const int nChunks = K >> 8;           // K is a multiple of 256
    stageB(bsh[0], BT, nBase, 0, K, tid);
    __syncthreads();

    v16bf a0 = load_fragA(A, K, rowA0, 0, lane);
    v16bf a1 = load_fragA(A, K, rowA1, 0, lane);

    for (int ch = 0; ch < nChunks; ++ch) {
        if (ch + 1 < nChunks)
            stageB(bsh[(ch + 1) & 1], BT, nBase, (ch + 1) << 8, K, tid);
        const __bf16* bb = bsh[ch & 1];
        const int kBase = ch << 8;
#pragma unroll
        for (int ks = 0; ks < 8; ++ks) {
            int kNext = kBase + (ks + 1) * 32;
            bool more = kNext < K;
            v16bf a0n = a0, a1n = a1;
            if (more) {                       // prefetch next A frags (reg double buffer)
                a0n = load_fragA(A, K, rowA0, kNext, lane);
                a1n = load_fragA(A, K, rowA1, kNext, lane);
            }
#pragma unroll
            for (int j = 0; j < 4; ++j) {
                v16bf b = load_fragB_lds(bb, j * 16 + lo, ks * 32 + hi * 8);
                acc[0][j] = __builtin_amdgcn_wmma_f32_16x16x32_bf16(
                    false, a0, false, b, (short)0, acc[0][j], false, false);
                acc[1][j] = __builtin_amdgcn_wmma_f32_16x16x32_bf16(
                    false, a1, false, b, (short)0, acc[1][j], false, false);
            }
            a0 = a0n; a1 = a1n;
        }
        __syncthreads();
    }

    // C layout: VGPR r -> M = r (lanes0-15) / 8+r (lanes16-31), N = lane&15
#pragma unroll
    for (int mi = 0; mi < 2; ++mi) {
#pragma unroll
        for (int j = 0; j < 4; ++j) {
            int col = nBase + j * 16 + lo;
            float bv = (EPI == 1) ? bias[col] : 0.0f;
#pragma unroll
            for (int r = 0; r < 8; ++r) {
                int row = mBase + mi * 16 + hi * 8 + r;
                float v = acc[mi][j][r];
                if (EPI == 0) {
                    Cf[(size_t)row * N + col] = v;
                } else {
                    v += bv;
                    v = v > 0.f ? v : 0.f;
                    Cb[(size_t)row * N + col] = f2bf(v);
                }
            }
        }
    }
}

// ---------------- persistent GRU scan (single WGP, Wh register-resident) ----------------
// 1024 threads. Thread t: c = t&255 (column within gate), s = t>>8 (k-slice of 64).
// Owns Wh[64s:64s+64, {c, 256+c, 512+c}] in 192 VGPRs. h broadcast via LDS.
// 2 barriers/step: gate threads reduce the 4 k-partials themselves (float4 LDS loads).
__global__ void __launch_bounds__(1024, 1)
k_gru_scan(const float* __restrict__ xg, const float* __restrict__ Wh,
           const float* __restrict__ bh, const float* __restrict__ h0,
           __bf16* __restrict__ ysrelu, int T) {
    __shared__ float hsh[DIM];
    __shared__ float part[3 * DIM * 4];

    int t = threadIdx.x;
    int c = t & (DIM - 1);
    int s = t >> 8;

    float wr[3][64];
#pragma unroll
    for (int g = 0; g < 3; ++g)
#pragma unroll
        for (int k = 0; k < 64; ++k)
            wr[g][k] = Wh[(size_t)(s * 64 + k) * (3 * DIM) + g * DIM + c];

    float bhr = bh[c], bhz = bh[DIM + c], bhn = bh[2 * DIM + c];   // loop-invariant

    if (t < DIM) hsh[t] = h0[t];
    __syncthreads();

    for (int step = 0; step < T; ++step) {
        const float* xgrow = xg + (size_t)step * (3 * DIM);
        __builtin_prefetch(xgrow + 3 * DIM, 0, 0);   // next step's xg row

        float a0 = 0.f, a1 = 0.f, a2 = 0.f;
#pragma unroll
        for (int k = 0; k < 64; ++k) {
            float hk = hsh[(s << 6) + k];            // same-address LDS broadcast
            a0 += wr[0][k] * hk;
            a1 += wr[1][k] * hk;
            a2 += wr[2][k] * hk;
        }
        part[(0 * DIM + c) * 4 + s] = a0;
        part[(1 * DIM + c) * 4 + s] = a1;
        part[(2 * DIM + c) * 4 + s] = a2;
        __syncthreads();

        if (t < DIM) {
            f32x4 pr = *(const f32x4*)&part[(0 * DIM + c) * 4];
            f32x4 pz = *(const f32x4*)&part[(1 * DIM + c) * 4];
            f32x4 pn = *(const f32x4*)&part[(2 * DIM + c) * 4];
            float hr = pr.x + pr.y + pr.z + pr.w + bhr;
            float hz = pz.x + pz.y + pz.z + pz.w + bhz;
            float hn = pn.x + pn.y + pn.z + pn.w + bhn;
            float xr = xgrow[c], xz = xgrow[DIM + c], xn = xgrow[2 * DIM + c];
            float r = 1.f / (1.f + __expf(-(xr + hr)));
            float z = 1.f / (1.f + __expf(-(xz + hz)));
            float nn = tanhf(xn + r * hn);
            float hnew = (1.f - z) * nn + z * hsh[c];
            hsh[c] = hnew;
            ysrelu[(size_t)step * DIM + c] = f2bf(hnew > 0.f ? hnew : 0.f);
        }
        __syncthreads();
    }
}

// ---------------- fused residual + bias + LayerNorm ----------------
// out = LN(base + add + bias) * gamma + beta ; optional bf16 mirror.
__global__ void __launch_bounds__(256)
k_ln(const float* __restrict__ base, const float* __restrict__ add,
     const float* __restrict__ bias, const float* __restrict__ gamma,
     const float* __restrict__ beta, float* __restrict__ outf,
     __bf16* __restrict__ outb) {
    __shared__ float red[DIM];
    int r = blockIdx.x, i = threadIdx.x;
    size_t off = (size_t)r * DIM + i;
    float x = base[off] + add[off] + bias[i];

    red[i] = x; __syncthreads();
#pragma unroll
    for (int sft = 128; sft > 0; sft >>= 1) {
        if (i < sft) red[i] += red[i + sft];
        __syncthreads();
    }
    float mu = red[0] * (1.0f / DIM);
    __syncthreads();
    float d = x - mu;
    red[i] = d * d; __syncthreads();
#pragma unroll
    for (int sft = 128; sft > 0; sft >>= 1) {
        if (i < sft) red[i] += red[i + sft];
        __syncthreads();
    }
    float var = red[0] * (1.0f / DIM);
    float y = d * rsqrtf(var + LN_EPS) * gamma[i] + beta[i];
    outf[off] = y;
    if (outb) outb[off] = f2bf(y);
}

extern "C" void kernel_launch(void* const* d_in, const int* in_sizes, int n_in,
                              void* d_out, int out_size, void* d_ws, size_t ws_size,
                              hipStream_t stream) {
    (void)in_sizes; (void)n_in; (void)out_size; (void)ws_size;
    const float* xs  = (const float*)d_in[0];
    const float* h0  = (const float*)d_in[1];
    const float* Wi  = (const float*)d_in[2];
    const float* Wh  = (const float*)d_in[3];
    const float* bh  = (const float*)d_in[4];
    const float* Wg  = (const float*)d_in[5];
    const float* bg  = (const float*)d_in[6];
    const float* g1  = (const float*)d_in[7];
    const float* be1 = (const float*)d_in[8];
    const float* W1  = (const float*)d_in[9];
    const float* b1  = (const float*)d_in[10];
    const float* W2  = (const float*)d_in[11];
    const float* b2  = (const float*)d_in[12];
    const float* g2  = (const float*)d_in[13];
    const float* be2 = (const float*)d_in[14];
    float* out = (float*)d_out;

    char* p = (char*)d_ws;
    auto alloc = [&](size_t bytes) { char* r = p; p += (bytes + 255) & ~(size_t)255; return r; };
    __bf16* xsb   = (__bf16*)alloc((size_t)TSTEPS * DIM * 2);        // bf16 xs
    __bf16* WiT   = (__bf16*)alloc((size_t)3 * DIM * DIM * 2);       // Wi^T
    __bf16* WgT   = (__bf16*)alloc((size_t)DIM * DIM * 2);           // Wg^T
    __bf16* W1T   = (__bf16*)alloc((size_t)FFN3 * DIM * 2);          // W1^T
    __bf16* W2T   = (__bf16*)alloc((size_t)DIM * FFN3 * 2);          // W2^T
    float*  xg    = (float*) alloc((size_t)TSTEPS * 3 * DIM * 4);    // xs@Wi
    __bf16* ysb   = (__bf16*)alloc((size_t)TSTEPS * DIM * 2);        // relu(ys) bf16
    float*  tmpwg = (float*) alloc((size_t)TSTEPS * DIM * 4);        // relu(ys)@Wg
    float*  ybuf  = (float*) alloc((size_t)TSTEPS * DIM * 4);        // y f32
    __bf16* yb    = (__bf16*)alloc((size_t)TSTEPS * DIM * 2);        // y bf16
    // xg is dead after the scan: reuse its 96 MB for FFN hidden + tmp2 (67+33.5 MB)
    __bf16* h1b   = (__bf16*)xg;
    float*  tmp2  = (float*)((char*)xg + (size_t)TSTEPS * FFN3 * 2);

    int n;
    n = TSTEPS * DIM;
    k_cvt_bf16<<<(n + 255) / 256, 256, 0, stream>>>(xs, xsb, n);
    n = DIM * 3 * DIM;
    k_transpose_bf16<<<(n + 255) / 256, 256, 0, stream>>>(Wi, WiT, DIM, 3 * DIM);
    n = DIM * DIM;
    k_transpose_bf16<<<(n + 255) / 256, 256, 0, stream>>>(Wg, WgT, DIM, DIM);
    n = DIM * FFN3;
    k_transpose_bf16<<<(n + 255) / 256, 256, 0, stream>>>(W1, W1T, DIM, FFN3);
    n = FFN3 * DIM;
    k_transpose_bf16<<<(n + 255) / 256, 256, 0, stream>>>(W2, W2T, FFN3, DIM);

    // xg = xs @ Wi  [T,768]
    k_gemm_bf16<0><<<dim3(TSTEPS / 256, (3 * DIM) / 64), 256, 0, stream>>>(
        xsb, WiT, nullptr, xg, nullptr, TSTEPS, 3 * DIM, DIM);
    // sequential GRU scan -> relu(ys) bf16
    k_gru_scan<<<1, 1024, 0, stream>>>(xg, Wh, bh, h0, ysb, TSTEPS);
    // tmpwg = relu(ys) @ Wg  [T,256]
    k_gemm_bf16<0><<<dim3(TSTEPS / 256, DIM / 64), 256, 0, stream>>>(
        ysb, WgT, nullptr, tmpwg, nullptr, TSTEPS, DIM, DIM);
    // y = LN(xs + tmpwg + bg)
    k_ln<<<TSTEPS, 256, 0, stream>>>(xs, tmpwg, bg, g1, be1, ybuf, yb);
    // h1 = relu(y @ W1 + b1)  [T,1024] bf16
    k_gemm_bf16<1><<<dim3(TSTEPS / 256, FFN3 / 64), 256, 0, stream>>>(
        yb, W1T, b1, nullptr, h1b, TSTEPS, FFN3, DIM);
    // tmp2 = h1 @ W2  [T,256]
    k_gemm_bf16<0><<<dim3(TSTEPS / 256, DIM / 64), 256, 0, stream>>>(
        h1b, W2T, nullptr, tmp2, nullptr, TSTEPS, DIM, FFN3);
    // out = LN(y + tmp2 + b2)
    k_ln<<<TSTEPS, 256, 0, stream>>>(ybuf, tmp2, b2, g2, be2, out, nullptr);
}
